// GCNLSTMRawSubGraphLossEmbedLoss_90692529422798
// MI455X (gfx1250) — matrix operand order repaired and checked by reading, not compiled, not run on hardware
//
#include <hip/hip_runtime.h>
#include <hip/hip_bf16.h>
#include <cstdint>

typedef __attribute__((ext_vector_type(16))) __bf16 v16bf;
typedef __attribute__((ext_vector_type(8)))  float  v8f;
typedef unsigned u32x4 __attribute__((ext_vector_type(4)));
typedef int      i32x8 __attribute__((ext_vector_type(8)));
typedef int      i32x4 __attribute__((ext_vector_type(4)));

#define HDIM 640
#define GDIM 2560
#define TSTEPS 2048
#define NNODES 2048
#define NEDGES 65536
#define RBLOCKS 40            // persistent recurrence workgroups (1 per WGP)
#define WROW 645              // effective LDS row stride: 640 + 1 pad DWORD per 128

// ---------------------------------------------------------------- utils
__global__ void k_zero(float* __restrict__ p, int n) {
  int i = blockIdx.x * blockDim.x + threadIdx.x;
  if (i < n) p[i] = 0.f;
}

__global__ void k_transpose(const float* __restrict__ W, float* __restrict__ Wt,
                            int rows, int cols) {
  int idx = blockIdx.x * blockDim.x + threadIdx.x;
  if (idx >= rows * cols) return;
  int i = idx / cols, o = idx - i * cols;
  Wt[(size_t)o * rows + i] = W[idx];
}

// ---------------------------------------------------------------- fragment loads
// A fragment (16x32 bf16): lane half = K-half selector.
__device__ __forceinline__ v16bf ldfragA(const float* __restrict__ r, int k0, int half) {
  v16bf a;
  int b0 = k0 + 16 * half, b1 = k0 + 8 + 16 * half;
#pragma unroll
  for (int e = 0; e < 8; ++e) a[e]     = (__bf16)r[b0 + e];
#pragma unroll
  for (int e = 0; e < 8; ++e) a[8 + e] = (__bf16)r[b1 + e];
  return a;
}
__device__ __forceinline__ v16bf ldfragA_g(const float* __restrict__ r, int k0, int half, int K) {
  v16bf a;
#pragma unroll
  for (int e = 0; e < 16; ++e) {
    int k = k0 + e + 16 * half;
    a[e] = (k < K) ? (__bf16)r[k] : (__bf16)0.0f;
  }
  return a;
}
// B fragment (32x16 bf16): blocked K layout.
__device__ __forceinline__ v16bf ldfragB(const float* __restrict__ r, int k0, int half) {
  v16bf b;
  int b0 = k0 + 16 * half;
#pragma unroll
  for (int e = 0; e < 16; ++e) b[e] = (__bf16)r[b0 + e];
  return b;
}
__device__ __forceinline__ v16bf ldfragB_g(const float* __restrict__ r, int k0, int half, int K) {
  v16bf b;
#pragma unroll
  for (int e = 0; e < 16; ++e) {
    int k = k0 + e + 16 * half;
    b[e] = (k < K) ? (__bf16)r[k] : (__bf16)0.0f;
  }
  return b;
}

// ---------------------------------------------------------------- WMMA GEMM (NT)
// C[M,N] = A[M,K] x B[N,K]^T (+ optional per-column biases).
// One wave owns a 64x64 supertile: 4 A frags x 4 B frags -> 16 WMMAs per
// 32-wide K step (~16 FLOP/B from L2, vs 4 for a single-tile wave).
// Out-of-range B columns use a clamped row pointer (never stored), so the
// K loop has no EXEC manipulation. M must be a multiple of 64.
__global__ void __launch_bounds__(256, 1)
k_gemm_nt(const float* __restrict__ A, const float* __restrict__ B,
          const float* __restrict__ bias0, const float* __restrict__ bias1,
          float* __restrict__ C, int M, int N, int K,
          int lda, int ldb, int ldc, int tn64)
{
  int wave = threadIdx.x >> 5;
  int lane = threadIdx.x & 31;
  int tile = blockIdx.x * 8 + wave;
  int tilesTotal = (M >> 6) * tn64;
  if (tile >= tilesTotal) return;                  // wave-uniform exit
  int tm = tile / tn64, tn = tile - tm * tn64;
  int m0 = tm * 64, n0 = tn * 64;

  int half = lane >> 4;
  int l15  = lane & 15;

  const float* Arow[4];
  const float* Brow[4];
  int colv[4];
#pragma unroll
  for (int i = 0; i < 4; ++i)
    Arow[i] = A + (size_t)(m0 + 16 * i + l15) * lda;
#pragma unroll
  for (int j = 0; j < 4; ++j) {
    int col = n0 + 16 * j + l15;
    colv[j] = col;
    int cc = (col < N) ? col : (N - 1);            // clamp: no branch in K loop
    Brow[j] = B + (size_t)cc * ldb;
  }

  v8f acc[4][4];
#pragma unroll
  for (int i = 0; i < 4; ++i)
#pragma unroll
    for (int j = 0; j < 4; ++j)
#pragma unroll
      for (int r = 0; r < 8; ++r) acc[i][j][r] = 0.f;

  int k0 = 0;
  for (; k0 + 32 <= K; k0 += 32) {
    v16bf bf[4];
#pragma unroll
    for (int j = 0; j < 4; ++j) bf[j] = ldfragB(Brow[j], k0, half);
#pragma unroll
    for (int i = 0; i < 4; ++i) {
      v16bf af = ldfragA(Arow[i], k0, half);
#pragma unroll
      for (int j = 0; j < 4; ++j)
        acc[i][j] = __builtin_amdgcn_wmma_f32_16x16x32_bf16(
            false, af, false, bf[j], (short)0, acc[i][j], false, false);
    }
  }
  if (k0 < K) {                                     // guarded K tail
    v16bf bf[4];
#pragma unroll
    for (int j = 0; j < 4; ++j) bf[j] = ldfragB_g(Brow[j], k0, half, K);
#pragma unroll
    for (int i = 0; i < 4; ++i) {
      v16bf af = ldfragA_g(Arow[i], k0, half, K);
#pragma unroll
      for (int j = 0; j < 4; ++j)
        acc[i][j] = __builtin_amdgcn_wmma_f32_16x16x32_bf16(
            false, af, false, bf[j], (short)0, acc[i][j], false, false);
    }
  }

#pragma unroll
  for (int j = 0; j < 4; ++j) {
    int col = colv[j];
    if (col < N) {
      float bias = 0.f;
      if (bias0) bias += bias0[col];
      if (bias1) bias += bias1[col];
#pragma unroll
      for (int i = 0; i < 4; ++i) {
#pragma unroll
        for (int r = 0; r < 8; ++r) {
          int m = m0 + 16 * i + r + 8 * half;       // C/D fragment layout
          C[(size_t)m * ldc + col] = acc[i][j][r] + bias;
        }
      }
    }
  }
}

// ---------------------------------------------------------------- LSTM recurrence
// Persistent kernel: 40 workgroups, each owns 16 h-lanes (64 gate rows).
// The 64x640 w_hh slice is staged into LDS by the Tensor Data Mover as a
// 3D tile (640 x 16 rows x 4 gate blocks, dim1-stride 640*640), with TDM
// padding (1 DWORD per 128) giving LDS offset row*645 + k + (k>>7) so a
// wave's 32 lanes (bank = row*5 + part mod 64) are conflict-free.
__global__ void __launch_bounds__(256)
k_lstm_recur(const float* __restrict__ xp,    // [T, 4H] (biases included)
             const float* __restrict__ w_hh,  // [4H, H]
             float* __restrict__ hs,          // [T, H]
             float* __restrict__ hbuf,        // [2, H] double buffer (zeroed)
             unsigned* __restrict__ bar)      // grid barrier counter (zeroed)
{
  extern __shared__ float smem[];
  float* sW = smem;                        // [64][WROW] (piecewise-padded)
  float* sH = smem + 64 * WROW + 8;        // [H]
  float* sG = sH + HDIM;                   // [64]

  int tid = threadIdx.x;
  int j0  = blockIdx.x * 16;

#if __has_builtin(__builtin_amdgcn_tensor_load_to_lds)
  if (tid < 32) {                          // wave 0 issues one TDM descriptor
    unsigned ldsb = (unsigned)(size_t)sW;  // low 32 bits of LDS aperture addr
    unsigned long long ga =
        (unsigned long long)(size_t)(w_hh + (size_t)j0 * HDIM);
    u32x4 g0; i32x8 g1; i32x4 g2, g3; i32x8 g4;
    g0[0] = 1u;                                        // count=1, user mode
    g0[1] = ldsb;                                      // lds_addr
    g0[2] = (unsigned)(ga & 0xFFFFFFFFu);              // global_addr lo
    g0[3] = (unsigned)((ga >> 32) & 0x01FFFFFFu) | (2u << 30);  // hi | type=2
    g1[0] = (2 << 16) | (1 << 20) | (6 << 22);         // 4B, pad_en, 128dw/1dw
    g1[1] = (int)((HDIM & 0xFFFFu) << 16);             // tensor_dim0=640 lo16
    g1[2] = (int)((GDIM & 0xFFFFu) << 16);             // dim0 hi | tensor_dim1 lo
    g1[3] = (int)((HDIM & 0xFFFFu) << 16);             // dim1 hi | tile_dim0=640
    g1[4] = 16 | (4 << 16);                            // tile_dim1=16, tile_dim2=4
    g1[5] = HDIM;                                      // dim0_stride=640 lo32
    g1[6] = (int)(((HDIM * HDIM) & 0xFFFFu) << 16);    // stride0 hi | stride1 lo16
    g1[7] = (int)((unsigned)(HDIM * HDIM) >> 16);      // stride1 hi32
    g2[0] = 16; g2[1] = 0; g2[2] = 0; g2[3] = 0;       // tensor_dim2 bound
    g3[0] = 0;  g3[1] = 0; g3[2] = 0; g3[3] = 0;
    g4[0] = 0; g4[1] = 0; g4[2] = 0; g4[3] = 0;        // VADDR4 slot: unused
    g4[4] = 0; g4[5] = 0; g4[6] = 0; g4[7] = 0;
    __builtin_amdgcn_tensor_load_to_lds(g0, g1, g2, g3, g4, 0);
    __builtin_amdgcn_s_wait_tensorcnt(0);
  }
#else
  for (int idx = tid; idx < 64 * HDIM; idx += 256) {
    int lr = idx / HDIM, k = idx - lr * HDIM;
    int q = lr >> 4, jj = lr & 15;
    sW[lr * WROW + k + (k >> 7)] = w_hh[(size_t)(q * HDIM + j0 + jj) * HDIM + k];
  }
#endif

  int lr = tid >> 2, part = tid & 3;
  int q = lr >> 4, jj = lr & 15;
  int grow = q * HDIM + j0 + jj;           // gate row in [0,4H)
  float c = 0.f;                           // cell state for tid<16 owners

  for (int t = 0; t < TSTEPS; ++t) {
    const float* hcur = hbuf + (t & 1) * HDIM;
    for (int k = tid; k < HDIM; k += 256) sH[k] = hcur[k];
    __syncthreads();

    float acc = 0.f;
    const float* wr = sW + lr * WROW;
#pragma unroll 8
    for (int k = part; k < HDIM; k += 4) acc += wr[k + (k >> 7)] * sH[k];
    acc += __shfl_xor(acc, 1);
    acc += __shfl_xor(acc, 2);
    if (part == 0) sG[lr] = acc + xp[(size_t)t * GDIM + grow];
    __syncthreads();

    if (tid < 16) {
      float gi = sG[tid], gf = sG[16 + tid], gg = sG[32 + tid], go = sG[48 + tid];
      float i_ = 1.f / (1.f + __expf(-gi));
      float f_ = 1.f / (1.f + __expf(-gf));
      float g_ = tanhf(gg);
      float o_ = 1.f / (1.f + __expf(-go));
      c = f_ * c + i_ * g_;
      float h = o_ * tanhf(c);
      hbuf[((t + 1) & 1) * HDIM + j0 + tid] = h;
      hs[(size_t)t * HDIM + j0 + tid] = h;
    }
    __syncthreads();

    // grid-wide step barrier (all 40 blocks co-resident, 1 per WGP)
    if (tid == 0) {
      __threadfence();
      atomicAdd(bar, 1u);
      unsigned target = (unsigned)(t + 1) * RBLOCKS;
      while (__hip_atomic_load(bar, __ATOMIC_ACQUIRE, __HIP_MEMORY_SCOPE_AGENT) < target)
        __builtin_amdgcn_s_sleep(1);
    }
    __syncthreads();
  }
}

// ---------------------------------------------------------------- GCN pieces
__global__ void k_deg(const int* __restrict__ ei, float* __restrict__ deg, int E) {
  int e = blockIdx.x * blockDim.x + threadIdx.x;
  if (e < E) atomicAdd(&deg[ei[E + e]], 1.0f);     // scatter at dst
}

__global__ void k_dinv(float* __restrict__ dnv, const float* __restrict__ deg, int n) {
  int i = blockIdx.x * blockDim.x + threadIdx.x;
  if (i < n) dnv[i] = rsqrtf(deg[i] + 1.0f);       // +1 self loop; always > 0
}

// out[n,f] = xw[n,f] * dinv[n]^2   (self-loop term; also initializes out)
__global__ void k_selfloop(const float* __restrict__ xw, const float* __restrict__ dnv,
                           float* __restrict__ out, int d) {
  int idx = blockIdx.x * blockDim.x + threadIdx.x;
  if (idx >= NNODES * d) return;
  int n = idx / d;
  float di = dnv[n];
  out[idx] = xw[idx] * di * di;
}

// one warp per edge; lanes sweep features (coalesced loads + atomics)
__global__ void k_edge(const int* __restrict__ ei, const float* __restrict__ xw,
                       const float* __restrict__ dnv, float* __restrict__ out,
                       int E, int d) {
  int warp = (blockIdx.x * blockDim.x + threadIdx.x) >> 5;
  int lane = threadIdx.x & 31;
  if (warp >= E) return;
  int s = ei[warp], t = ei[E + warp];
  float nrm = dnv[s] * dnv[t];
  const float* xs = xw + (size_t)s * d;
  float* od = out + (size_t)t * d;
  for (int f = lane; f < d; f += 32) atomicAdd(&od[f], xs[f] * nrm);
}

__global__ void k_bias_lrelu(float* __restrict__ x, const float* __restrict__ b,
                             int tot, int d) {
  int idx = blockIdx.x * blockDim.x + threadIdx.x;
  if (idx >= tot) return;
  float v = x[idx] + b[idx % d];
  x[idx] = (v >= 0.f) ? v : 0.01f * v;
}

// training-mode batch norm, no affine: one block per feature column
__global__ void k_bn(float* __restrict__ x, int n, int d) {
  __shared__ float red[256];
  int f = blockIdx.x, tid = threadIdx.x;
  float s = 0.f, s2 = 0.f;
  for (int r = tid; r < n; r += 256) {
    float v = x[(size_t)r * d + f];
    s += v; s2 += v * v;
  }
  red[tid] = s; __syncthreads();
  for (int o = 128; o; o >>= 1) { if (tid < o) red[tid] += red[tid + o]; __syncthreads(); }
  float mu = red[0] / (float)n;
  __syncthreads();
  red[tid] = s2; __syncthreads();
  for (int o = 128; o; o >>= 1) { if (tid < o) red[tid] += red[tid + o]; __syncthreads(); }
  float var = red[0] / (float)n - mu * mu;
  float is = rsqrtf(var + 1e-5f);
  for (int r = tid; r < n; r += 256) {
    size_t i = (size_t)r * d + f;
    x[i] = (x[i] - mu) * is;
  }
}

// global_add_pool over contiguous 128-node blocks: emb[16,50]
__global__ void k_pool(const float* __restrict__ x, float* __restrict__ emb) {
  int idx = blockIdx.x * blockDim.x + threadIdx.x;
  if (idx >= 16 * 50) return;
  int b = idx / 50, f = idx - b * 50;
  float s = 0.f;
  for (int r = 0; r < 128; ++r) s += x[(size_t)(b * 128 + r) * 50 + f];
  emb[idx] = s;
}

// tiny FC head, single block; also writes emb into d_out[16..]
__global__ void __launch_bounds__(512)
k_fc(const float* __restrict__ emb,
     const float* __restrict__ w1, const float* __restrict__ b1,
     const float* __restrict__ w2, const float* __restrict__ b2,
     const float* __restrict__ w3, const float* __restrict__ b3,
     float* __restrict__ out) {
  __shared__ float se[800];
  __shared__ float t1[16 * 32];
  __shared__ float t2[16 * 16];
  int tid = threadIdx.x;
  for (int i = tid; i < 800; i += 512) { se[i] = emb[i]; out[16 + i] = emb[i]; }
  __syncthreads();
  { int r = tid >> 5, c = tid & 31;               // 512 = 16x32
    float s = b1[c];
    for (int k = 0; k < 50; ++k) s += se[r * 50 + k] * w1[k * 32 + c];
    t1[tid] = s; }
  __syncthreads();
  if (tid < 256) { int r = tid >> 4, c = tid & 15;
    float s = b2[c];
    for (int k = 0; k < 32; ++k) s += t1[r * 32 + k] * w2[k * 16 + c];
    t2[tid] = s; }
  __syncthreads();
  if (tid < 16) {
    float s = b3[0];
    for (int k = 0; k < 16; ++k) s += t2[tid * 16 + k] * w3[k];
    out[tid] = s;
  }
}

// ---------------------------------------------------------------- launch
extern "C" void kernel_launch(void* const* d_in, const int* in_sizes, int n_in,
                              void* d_out, int out_size, void* d_ws, size_t ws_size,
                              hipStream_t stream) {
  (void)in_sizes; (void)n_in; (void)out_size; (void)ws_size;

  const float* x_in = (const float*)d_in[0];
  const int*   ei   = (const int*)  d_in[1];
  const float* w_ih[3] = {(const float*)d_in[2], (const float*)d_in[6], (const float*)d_in[10]};
  const float* w_hh[3] = {(const float*)d_in[3], (const float*)d_in[7], (const float*)d_in[11]};
  const float* b_ih[3] = {(const float*)d_in[4], (const float*)d_in[8], (const float*)d_in[12]};
  const float* b_hh[3] = {(const float*)d_in[5], (const float*)d_in[9], (const float*)d_in[13]};
  const float* gw[4] = {(const float*)d_in[14], (const float*)d_in[16], (const float*)d_in[18], (const float*)d_in[20]};
  const float* gb[4] = {(const float*)d_in[15], (const float*)d_in[17], (const float*)d_in[19], (const float*)d_in[21]};
  const float* fw[3] = {(const float*)d_in[22], (const float*)d_in[24], (const float*)d_in[26]};
  const float* fb[3] = {(const float*)d_in[23], (const float*)d_in[25], (const float*)d_in[27]};
  float* out = (float*)d_out;

  float* ws = (float*)d_ws;
  size_t o = 0;
  float* xp   = ws;        o += (size_t)TSTEPS * GDIM;     // 20.97 MB
  float* hsA  = ws + o;    o += (size_t)TSTEPS * HDIM;
  float* hsB  = ws + o;    o += (size_t)TSTEPS * HDIM;
  float* hbuf = ws + o;    o += 2 * HDIM;
  unsigned* bar = (unsigned*)(ws + o); o += 32;
  float* xw   = ws + o;    o += (size_t)NNODES * 320;
  float* gA   = ws + o;    o += (size_t)NNODES * 320;
  float* gB   = ws + o;    o += (size_t)NNODES * 320;
  float* Wt   = ws + o;    o += (size_t)640 * 320;
  float* deg  = ws + o;    o += NNODES;
  float* dnv  = ws + o;    o += NNODES;
  float* emb  = ws + o;    o += 800;

  const int smem = (64 * WROW + 8 + HDIM + 64) * 4;   // ~168 KB (< 320KB/WGP)
  (void)hipFuncSetAttribute((const void*)k_lstm_recur,
                            hipFuncAttributeMaxDynamicSharedMemorySize, smem);

  auto gemm = [&](const float* A, const float* B, const float* bz0, const float* bz1,
                  float* C, int M, int N, int K) {
    int tn64   = (N + 63) / 64;
    int tiles  = (M / 64) * tn64;
    int blocks = (tiles + 7) / 8;
    k_gemm_nt<<<blocks, 256, 0, stream>>>(A, B, bz0, bz1, C, M, N, K, K, K, N, tn64);
  };

  // ---- 3-layer LSTM: WMMA projection + persistent recurrence
  const float* lin = x_in;
  int lk = 8500;
  float* hs_out[3] = {hsA, hsB, hsA};
  for (int l = 0; l < 3; ++l) {
    gemm(lin, w_ih[l], b_ih[l], b_hh[l], xp, TSTEPS, GDIM, lk);
    k_zero<<<(2 * HDIM + 32 + 255) / 256, 256, 0, stream>>>(hbuf, 2 * HDIM + 32);
    k_lstm_recur<<<RBLOCKS, 256, smem, stream>>>(xp, w_hh[l], hs_out[l], hbuf, bar);
    lin = hs_out[l];
    lk = HDIM;
  }

  // ---- degree normalization (shared by all 4 GCN layers)
  k_zero<<<(NNODES + 255) / 256, 256, 0, stream>>>(deg, NNODES);
  k_deg<<<NEDGES / 256, 256, 0, stream>>>(ei, deg, NEDGES);
  k_dinv<<<(NNODES + 255) / 256, 256, 0, stream>>>(dnv, deg, NNODES);

  // ---- 4 GCN layers
  int din[4]  = {640, 320, 180, 90};
  int dou[4]  = {320, 180, 90, 50};
  const float* xcur = hsA;
  float* gobuf[2] = {gA, gB};
  for (int l = 0; l < 4; ++l) {
    int di = din[l], dn = dou[l];
    k_transpose<<<(di * dn + 255) / 256, 256, 0, stream>>>(gw[l], Wt, di, dn);
    gemm(xcur, Wt, nullptr, nullptr, xw, NNODES, dn, di);
    float* go = gobuf[l & 1];
    int tot = NNODES * dn;
    k_selfloop<<<(tot + 255) / 256, 256, 0, stream>>>(xw, dnv, go, dn);
    k_edge<<<NEDGES / 8, 256, 0, stream>>>(ei, xw, dnv, go, NEDGES, dn);
    k_bias_lrelu<<<(tot + 255) / 256, 256, 0, stream>>>(go, gb[l], tot, dn);
    k_bn<<<dn, 256, 0, stream>>>(go, NNODES, dn);
    xcur = go;
  }

  // ---- pool + FC head (+ emits y[16] then emb[800] into d_out)
  k_pool<<<4, 256, 0, stream>>>(xcur, emb);
  k_fc<<<1, 512, 0, stream>>>(emb, fw[0], fb[0], fw[1], fb[1], fw[2], fb[2], out);
}
